// DecoderNetwork_16973710754224
// MI455X (gfx1250) — compile-verified, matrix operand order checked by reading, not run on hardware
//
#include <hip/hip_runtime.h>

// ---------------------------------------------------------------------------
// DecoderNetwork on MI455X (gfx1250): bf16 WMMA GEMMs (f32 accumulate),
// L2-resident prepacked weights, async global->LDS double-buffered B tiles,
// bf16 activation hand-off between kernels.
// B=256, L=128, H=1024, V=512, NL=2.
// ---------------------------------------------------------------------------

#define Bz 256
#define Lz 128
#define Hz 1024
#define Vz 512

typedef __bf16 bf16_t;
typedef __attribute__((ext_vector_type(16))) bf16_t v16bf;
typedef __attribute__((ext_vector_type(8)))  float  v8f;
typedef __attribute__((ext_vector_type(8)))  unsigned short ushort8;

union frag_u { v16bf bf; ushort8 us[2]; };

__device__ __forceinline__ unsigned short f2bf(float f) {
  union { float f; unsigned u; } v; v.f = f;
  unsigned r = v.u + 0x7FFFu + ((v.u >> 16) & 1u);   // round-to-nearest-even
  return (unsigned short)(r >> 16);
}
__device__ __forceinline__ float bf2f(unsigned short h) {
  union { unsigned u; float f; } v; v.u = ((unsigned)h) << 16;
  return v.f;
}
__device__ __forceinline__ float sigm(float x) { return 1.f / (1.f + expf(-x)); }

// Async DMA: 16 bytes per lane, global -> LDS, tracked by ASYNCcnt.
__device__ __forceinline__ void async_b128(unsigned lds_off, const void* gptr) {
  asm volatile("global_load_async_to_lds_b128 %0, %1, off"
               :: "v"(lds_off), "v"(gptr) : "memory");
}

// 4 accumulating WMMAs against the 4 B tiles of one staged LDS slab.
__device__ __forceinline__ void wmma4(const unsigned short* sb, int lane,
                                      const frag_u& a,
                                      v8f& acc0, v8f& acc1, v8f& acc2, v8f& acc3) {
  v16bf b0 = *(const v16bf*)(sb + 0 * 512 + lane * 16);
  v16bf b1 = *(const v16bf*)(sb + 1 * 512 + lane * 16);
  v16bf b2 = *(const v16bf*)(sb + 2 * 512 + lane * 16);
  v16bf b3 = *(const v16bf*)(sb + 3 * 512 + lane * 16);
  acc0 = __builtin_amdgcn_wmma_f32_16x16x32_bf16(false, a.bf, false, b0, (short)0, acc0, false, false);
  acc1 = __builtin_amdgcn_wmma_f32_16x16x32_bf16(false, a.bf, false, b1, (short)0, acc1, false, false);
  acc2 = __builtin_amdgcn_wmma_f32_16x16x32_bf16(false, a.bf, false, b2, (short)0, acc2, false, false);
  acc3 = __builtin_amdgcn_wmma_f32_16x16x32_bf16(false, a.bf, false, b3, (short)0, acc3, false, false);
}

// ---------------------------------------------------------------------------
// Prepack a logical B matrix B[k][n] = W[n][k] (optionally split over two
// source matrices along K, optional single extra row at n==Nvalid) into
// WMMA-fragment order: [nt][kt][lane][e], 512 bf16 per 32(K)x16(N) tile.
// Fragment layout per CDNA5 ISA (16-bit, 16x32/32x16, wave32):
//   lane = 16*khalfSel + n16 ; elements 0..7 -> K = khalf*8 + 0..7,
//   elements 8..15 -> K = 16 + khalf*8 + 0..7.
// ---------------------------------------------------------------------------
__global__ void k_prepack(const float* __restrict__ W1, const float* __restrict__ W2,
                          int ksplit, int K, int Ntiles, int Nvalid,
                          const float* __restrict__ extraRow,
                          unsigned short* __restrict__ out)
{
  int Ktiles = K >> 5;
  long long total = (long long)Ntiles * Ktiles * 512;
  long long idx = (long long)blockIdx.x * blockDim.x + threadIdx.x;
  if (idx >= total) return;
  int r = (int)(idx & 511);
  long long tile = idx >> 9;
  int kt = (int)(tile % Ktiles);
  int nt = (int)(tile / Ktiles);
  int lane = r >> 4, e = r & 15;
  int n = nt * 16 + (lane & 15);
  int khalf = (lane >> 4) * 8;
  int k = kt * 32 + ((e < 8) ? (khalf + e) : (16 + khalf + (e - 8)));
  float w = 0.f;
  if (n < Nvalid) {
    w = (k < ksplit) ? W1[(long long)n * ksplit + k]
                     : W2[(long long)n * (K - ksplit) + (k - ksplit)];
  } else if (extraRow != nullptr && n == Nvalid) {
    w = extraRow[k];
  }
  out[idx] = f2bf(w);
}

// ---------------------------------------------------------------------------
// bf16 WMMA GEMM: C[M,N] = A[M,K] * Bpack + bias, optional tanh.
// Block = 256 threads (8 waves), each wave a 16(M) x 64(N) strip -> block
// covers 128 rows; the 4 B tiles per K-chunk are shared by all 8 waves via
// LDS, filled with double-buffered GLOBAL_LOAD_ASYNC_TO_LDS_B128 (each wave
// copies one 512B chunk of the 4KB slab per K-chunk; s_wait_asynccnt +
// workgroup barrier before use). Last K-chunk peeled so the hot loop has no
// branches around the async issue/wait. grid = (M/128, Ntiles/4).
// ---------------------------------------------------------------------------
__global__ void __launch_bounds__(256)
k_gemm(const unsigned short* __restrict__ A, int lda,
       const unsigned short* __restrict__ Bp, int Ktiles,
       float* __restrict__ Cf, unsigned short* __restrict__ Cbf,
       int ldc, const float* __restrict__ bias, int actTanh)
{
  __shared__ unsigned short shB[2][4 * 512];   // 2 x 4KB B slabs

  int lane = threadIdx.x & 31;
  int wave = threadIdx.x >> 5;       // 0..7
  int m0  = blockIdx.x * 128 + wave * 16;
  int nt0 = blockIdx.y * 4;

  int mrow = m0 + (lane & 15);
  const unsigned short* aRow = A + (long long)mrow * lda + ((lane >> 4) * 8);

  // This wave's 512B async-copy chunk: tile ti (0..3), half hf (0..1).
  int ti = wave >> 1, hf = wave & 1;
  const unsigned short* gChunk =
      Bp + ((long long)(nt0 + ti) * Ktiles) * 512 + hf * 256 + lane * 8;
  unsigned ldsChunk0 = (unsigned)(unsigned long long)(const void*)
      &shB[0][ti * 512 + hf * 256 + lane * 8];
  unsigned ldsChunk1 = (unsigned)(unsigned long long)(const void*)
      &shB[1][ti * 512 + hf * 256 + lane * 8];

  v8f acc0 = {}, acc1 = {}, acc2 = {}, acc3 = {};

  // Prologue: stage K-chunk 0.
  async_b128(ldsChunk0, gChunk);

  int kt = 0;
  for (; kt < Ktiles - 1; ++kt) {
    // Stage next slab while computing this one (unconditional in hot loop).
    async_b128(((kt + 1) & 1) ? ldsChunk1 : ldsChunk0,
               gChunk + (long long)(kt + 1) * 512);

    // A fragment (unique per wave) straight from global/L2.
    frag_u a;
    a.us[0] = *(const ushort8*)(aRow + kt * 32);
    a.us[1] = *(const ushort8*)(aRow + kt * 32 + 16);

    // My copy for chunk kt has landed (kt+1 still in flight)...
    asm volatile("s_wait_asynccnt 0x1" ::: "memory");
    __syncthreads();   // ...and so has everyone else's.

    wmma4(shB[kt & 1], lane, a, acc0, acc1, acc2, acc3);

    __syncthreads();   // slab kt free for the overwrite issued next iteration
  }

  // Peeled final K-chunk: nothing left to stage.
  {
    frag_u a;
    a.us[0] = *(const ushort8*)(aRow + kt * 32);
    a.us[1] = *(const ushort8*)(aRow + kt * 32 + 16);
    asm volatile("s_wait_asynccnt 0x0" ::: "memory");
    __syncthreads();
    wmma4(shB[kt & 1], lane, a, acc0, acc1, acc2, acc3);
  }

  // C fragment layout: VGPR r -> row = r + (lane>=16 ? 8 : 0), col = lane&15.
  int rowOff = (lane >> 4) * 8;
  #pragma unroll
  for (int t = 0; t < 4; ++t) {
    v8f av = (t == 0) ? acc0 : (t == 1) ? acc1 : (t == 2) ? acc2 : acc3;
    int col = (nt0 + t) * 16 + (lane & 15);
    float bval = bias ? bias[col] : 0.f;
    #pragma unroll
    for (int r = 0; r < 8; ++r) {
      int row = m0 + rowOff + r;
      float v = av[r] + bval;
      if (actTanh) v = tanhf(v);
      if (Cf) Cf[(long long)row * ldc + col] = v;
      else    Cbf[(long long)row * ldc + col] = f2bf(v);
    }
  }
}

// ---------------------------------------------------------------------------
// Elementwise LSTM cell: gates [256,4096] fp32 (i,f,g,o), c state fp32 in/out,
// h written as bf16 to one or two destinations (next GEMM's A slots).
// ---------------------------------------------------------------------------
__global__ void k_lstm_cell(const float* __restrict__ gates, float* __restrict__ c,
                            unsigned short* __restrict__ hA, int strideA,
                            unsigned short* __restrict__ hB, int strideB)
{
  int idx = blockIdx.x * blockDim.x + threadIdx.x;   // Bz*Hz
  int b = idx >> 10, h = idx & 1023;
  const float* g = gates + (long long)b * 4096;
  float i  = sigm(g[h]);
  float f  = sigm(g[1024 + h]);
  float gg = tanhf(g[2048 + h]);
  float o  = sigm(g[3072 + h]);
  float cn = f * c[idx] + i * gg;
  c[idx] = cn;
  float hn = o * tanhf(cn);
  unsigned short hb = f2bf(hn);
  hA[(long long)b * strideA + h] = hb;
  if (hB) hB[(long long)b * strideB + h] = hb;
}

// fp32 -> bf16 bulk convert
__global__ void k_f32_to_bf16(const float* __restrict__ src,
                              unsigned short* __restrict__ dst, long long n)
{
  long long i = (long long)blockIdx.x * blockDim.x + threadIdx.x;
  if (i < n) dst[i] = f2bf(src[i]);
}

// ctx0 = mean over L of encoder_outputs -> bf16 into xcat0 ctx slot
__global__ void k_ctx0(const float* __restrict__ enc, unsigned short* __restrict__ xcat0)
{
  int idx = blockIdx.x * blockDim.x + threadIdx.x;   // Bz*Hz
  int b = idx >> 10, h = idx & 1023;
  const float* e = enc + (long long)b * Lz * Hz + h;
  float s = 0.f;
  for (int l = 0; l < Lz; ++l) s += e[(long long)l * Hz];
  xcat0[(long long)b * 3072 + 1024 + h] = f2bf(s * (1.f / Lz));
}

// init recurrent state: h -> bf16 A-slots, c -> fp32 state buffers
__global__ void k_init(const float* __restrict__ h0in, const float* __restrict__ c0in,
                       unsigned short* __restrict__ xcat0, unsigned short* __restrict__ xcat1,
                       float* __restrict__ c0s, float* __restrict__ c1s)
{
  int idx = blockIdx.x * blockDim.x + threadIdx.x;   // Bz*Hz
  int b = idx >> 10, h = idx & 1023;
  xcat0[(long long)b * 3072 + 2048 + h] = f2bf(h0in[idx]);             // layer 0 h
  xcat1[(long long)b * 2048 + 1024 + h] = f2bf(h0in[Bz * Hz + idx]);   // layer 1 h
  c0s[idx] = c0in[idx];
  c1s[idx] = c0in[Bz * Hz + idx];
}

__global__ void k_bias2(const float* __restrict__ x, const float* __restrict__ y,
                        float* __restrict__ o, int n)
{
  int i = blockIdx.x * blockDim.x + threadIdx.x;
  if (i < n) o[i] = x[i] + y[i];
}

__global__ void k_biasA(const float* __restrict__ actor_b, const float* __restrict__ critic_b,
                        float* __restrict__ o)
{
  int i = blockIdx.x * blockDim.x + threadIdx.x;   // 576
  if (i < 576) o[i] = (i < Vz) ? actor_b[i] : ((i == Vz) ? critic_b[0] : 0.f);
}

__global__ void k_action(const int* __restrict__ actions, float* __restrict__ out, int n)
{
  int i = blockIdx.x * blockDim.x + threadIdx.x;
  if (i < n) out[i] = (float)actions[i];
}

// teacher-forced embedding gather for step t -> bf16 into xcat0[:,0:1024]
__global__ void k_embed(const int* __restrict__ actions, const float* __restrict__ embedding,
                        unsigned short* __restrict__ xcat0, int t)
{
  int idx = blockIdx.x * blockDim.x + threadIdx.x;   // Bz*Hz
  int b = idx >> 10, h = idx & 1023;
  int id = (t == 0) ? 0 : actions[b * Lz + (t - 1)];
  xcat0[(long long)b * 3072 + h] = f2bf(embedding[(long long)id * Hz + h]);
}

// ---------------------------------------------------------------------------
// Attention: per-batch block. scores = out . keysW[b,l,:], softmax over L,
// ctx_att = attn . enc[b]; ctx_att (bf16) -> xcatC[:,1024:2048].
// ---------------------------------------------------------------------------
__global__ void k_attention(const unsigned short* __restrict__ xcatC,
                            const unsigned short* __restrict__ keysW,
                            const unsigned short* __restrict__ encbf,
                            unsigned short* __restrict__ ctxDst)  // = xcatC + 1024
{
  __shared__ float outv[Hz];
  __shared__ float attn[Lz];
  __shared__ float red[Lz];
  int b = blockIdx.x;
  int tid = threadIdx.x;  // 128
  for (int h = tid; h < Hz; h += 128)
    outv[h] = bf2f(xcatC[(long long)b * 2048 + h]);
  __syncthreads();

  const unsigned short* krow = keysW + ((long long)(b * Lz + tid)) * Hz;
  float acc = 0.f;
  for (int h = 0; h < Hz; ++h) acc += outv[h] * bf2f(krow[h]);
  attn[tid] = acc;
  __syncthreads();

  red[tid] = attn[tid]; __syncthreads();
  for (int s = 64; s > 0; s >>= 1) { if (tid < s) red[tid] = fmaxf(red[tid], red[tid + s]); __syncthreads(); }
  float mx = red[0]; __syncthreads();
  float e = expf(attn[tid] - mx);
  red[tid] = e; __syncthreads();
  for (int s = 64; s > 0; s >>= 1) { if (tid < s) red[tid] += red[tid + s]; __syncthreads(); }
  float inv = 1.f / red[0]; __syncthreads();
  attn[tid] = e * inv;
  __syncthreads();

  float a8[8] = {0.f, 0.f, 0.f, 0.f, 0.f, 0.f, 0.f, 0.f};
  for (int l = 0; l < Lz; ++l) {
    float w = attn[l];
    const unsigned short* er = encbf + ((long long)(b * Lz + l)) * Hz;
    #pragma unroll
    for (int j = 0; j < 8; ++j) a8[j] += w * bf2f(er[tid + j * 128]);
  }
  #pragma unroll
  for (int j = 0; j < 8; ++j)
    ctxDst[(long long)b * 2048 + tid + j * 128] = f2bf(a8[j]);
}

// ---------------------------------------------------------------------------
// Head: softmax over V=512 actor logits -> pi, value from fused critic col 512.
// ---------------------------------------------------------------------------
__global__ void k_head(const float* __restrict__ actbuf, float* __restrict__ logits,
                       float* __restrict__ values, int t)
{
  __shared__ float red[128];
  int b = blockIdx.x, tid = threadIdx.x;  // 128
  const float* row = actbuf + (long long)b * 576;
  float v4[4];
  float mx = -1e30f;
  #pragma unroll
  for (int j = 0; j < 4; ++j) { v4[j] = row[tid + j * 128]; mx = fmaxf(mx, v4[j]); }
  red[tid] = mx; __syncthreads();
  for (int s = 64; s > 0; s >>= 1) { if (tid < s) red[tid] = fmaxf(red[tid], red[tid + s]); __syncthreads(); }
  mx = red[0]; __syncthreads();
  float sum = 0.f;
  #pragma unroll
  for (int j = 0; j < 4; ++j) { v4[j] = expf(v4[j] - mx); sum += v4[j]; }
  red[tid] = sum; __syncthreads();
  for (int s = 64; s > 0; s >>= 1) { if (tid < s) red[tid] += red[tid + s]; __syncthreads(); }
  float inv = 1.f / red[0];
  float* dst = logits + ((long long)(b * Lz + t)) * Vz;
  #pragma unroll
  for (int j = 0; j < 4; ++j) dst[tid + j * 128] = v4[j] * inv;
  if (tid == 0) values[b * Lz + t] = row[Vz];
}

// ---------------------------------------------------------------------------
extern "C" void kernel_launch(void* const* d_in, const int* in_sizes, int n_in,
                              void* d_out, int out_size, void* d_ws, size_t ws_size,
                              hipStream_t stream)
{
  const float* enc       = (const float*)d_in[0];
  const float* h0in      = (const float*)d_in[1];
  const float* c0in      = (const float*)d_in[2];
  const int*   actions   = (const int*)d_in[3];
  const float* embedding = (const float*)d_in[4];
  const float* w_ih0     = (const float*)d_in[5];
  const float* w_hh0     = (const float*)d_in[6];
  const float* b_ih0     = (const float*)d_in[7];
  const float* b_hh0     = (const float*)d_in[8];
  const float* w_ih1     = (const float*)d_in[9];
  const float* w_hh1     = (const float*)d_in[10];
  const float* b_ih1     = (const float*)d_in[11];
  const float* b_hh1     = (const float*)d_in[12];
  const float* attn_w    = (const float*)d_in[13];
  const float* concat_w  = (const float*)d_in[14];
  const float* actor_w   = (const float*)d_in[15];
  const float* actor_b   = (const float*)d_in[16];
  const float* critic_w  = (const float*)d_in[17];
  const float* critic_b  = (const float*)d_in[18];
  (void)in_sizes; (void)n_in; (void)out_size; (void)ws_size;

  float* out_action = (float*)d_out;
  float* out_logits = out_action + (size_t)Bz * Lz;
  float* out_values = out_logits + (size_t)Bz * Lz * Vz;

  char* p = (char*)d_ws;
  auto take = [&](size_t bytes) -> char* {
    char* r = p; p += (bytes + 255) & ~(size_t)255; return r;
  };
  // Prepacked bf16 weights (L2-resident, ~49 MB)
  unsigned short* W0p  = (unsigned short*)take((size_t)256 * 96 * 512 * 2); // [4096n,3072k]
  unsigned short* W1p  = (unsigned short*)take((size_t)256 * 64 * 512 * 2); // [4096n,2048k]
  unsigned short* Wcp  = (unsigned short*)take((size_t)64  * 64 * 512 * 2); // [1024n,2048k]
  unsigned short* Wap  = (unsigned short*)take((size_t)36  * 32 * 512 * 2); // [576n,1024k] (+critic)
  unsigned short* Watt = (unsigned short*)take((size_t)64  * 32 * 512 * 2); // [1024n,1024k]
  unsigned short* keysW = (unsigned short*)take((size_t)Bz * Lz * Hz * 2);
  unsigned short* encbf = (unsigned short*)take((size_t)Bz * Lz * Hz * 2);
  unsigned short* xcat0 = (unsigned short*)take((size_t)Bz * 3072 * 2);     // [emb|ctx|h0]
  unsigned short* xcat1 = (unsigned short*)take((size_t)Bz * 2048 * 2);     // [h0n|h1]
  unsigned short* xcatC = (unsigned short*)take((size_t)Bz * 2048 * 2);     // [out|ctx_att]
  float* gates  = (float*)take((size_t)Bz * 4096 * 4);
  float* c0s    = (float*)take((size_t)Bz * Hz * 4);
  float* c1s    = (float*)take((size_t)Bz * Hz * 4);
  float* actbuf = (float*)take((size_t)Bz * 576 * 4);
  float* bias0  = (float*)take((size_t)4096 * 4);
  float* bias1  = (float*)take((size_t)4096 * 4);
  float* biasA  = (float*)take((size_t)576 * 4);

  dim3 blk256(256), blkG(256);
  long long nBH = (long long)Bz * Hz;

  // ---- one-time setup (re-run every call; deterministic) ----
  {
    long long n;
    n = 256LL * 96 * 512;
    k_prepack<<<dim3((unsigned)((n + 255) / 256)), blk256, 0, stream>>>(w_ih0, w_hh0, 2048, 3072, 256, 4096, nullptr, W0p);
    n = 256LL * 64 * 512;
    k_prepack<<<dim3((unsigned)((n + 255) / 256)), blk256, 0, stream>>>(w_ih1, w_hh1, 1024, 2048, 256, 4096, nullptr, W1p);
    n = 64LL * 64 * 512;
    k_prepack<<<dim3((unsigned)((n + 255) / 256)), blk256, 0, stream>>>(concat_w, nullptr, 2048, 2048, 64, 1024, nullptr, Wcp);
    n = 36LL * 32 * 512;
    k_prepack<<<dim3((unsigned)((n + 255) / 256)), blk256, 0, stream>>>(actor_w, nullptr, 1024, 1024, 36, 512, critic_w, Wap);
    n = 64LL * 32 * 512;
    k_prepack<<<dim3((unsigned)((n + 255) / 256)), blk256, 0, stream>>>(attn_w, nullptr, 1024, 1024, 64, 1024, nullptr, Watt);
  }
  k_bias2<<<dim3(4096 / 256), blk256, 0, stream>>>(b_ih0, b_hh0, bias0, 4096);
  k_bias2<<<dim3(4096 / 256), blk256, 0, stream>>>(b_ih1, b_hh1, bias1, 4096);
  k_biasA<<<dim3(3), blk256, 0, stream>>>(actor_b, critic_b, biasA);
  k_action<<<dim3((Bz * Lz + 255) / 256), blk256, 0, stream>>>(actions, out_action, Bz * Lz);

  {
    long long n = (long long)Bz * Lz * Hz;
    k_f32_to_bf16<<<dim3((unsigned)((n + 255) / 256)), blk256, 0, stream>>>(enc, encbf, n);
  }
  k_ctx0<<<dim3((unsigned)(nBH / 256)), blk256, 0, stream>>>(enc, xcat0);
  k_init<<<dim3((unsigned)(nBH / 256)), blk256, 0, stream>>>(h0in, c0in, xcat0, xcat1, c0s, c1s);

  // keysW = enc_bf16 [32768,1024] x attn_w^T -> bf16 [32768,1024]  (WMMA)
  k_gemm<<<dim3(Bz * Lz / 128, 16), blkG, 0, stream>>>(encbf, 1024, Watt, 32,
                                                       nullptr, keysW, 1024, nullptr, 0);

  // ---- recurrent decode, 128 steps ----
  for (int t = 0; t < Lz; ++t) {
    k_embed<<<dim3((unsigned)(nBH / 256)), blk256, 0, stream>>>(actions, embedding, xcat0, t);
    // LSTM layer 0: gates = [emb|ctx|h0] x W0 + b0
    k_gemm<<<dim3(2, 64), blkG, 0, stream>>>(xcat0, 3072, W0p, 96, gates, nullptr, 4096, bias0, 0);
    k_lstm_cell<<<dim3((unsigned)(nBH / 256)), blk256, 0, stream>>>(gates, c0s,
                                                                    xcat1, 2048,        // h0n -> G1 input
                                                                    xcat0 + 2048, 3072);// h0n -> next-step G0
    // LSTM layer 1: gates = [h0n|h1] x W1 + b1
    k_gemm<<<dim3(2, 64), blkG, 0, stream>>>(xcat1, 2048, W1p, 64, gates, nullptr, 4096, bias1, 0);
    k_lstm_cell<<<dim3((unsigned)(nBH / 256)), blk256, 0, stream>>>(gates, c1s,
                                                                    xcatC, 2048,        // out -> attn/concat
                                                                    xcat1 + 1024, 2048);// h1 -> next-step G1
    // attention -> ctx_att into xcatC[:,1024:2048]
    k_attention<<<dim3(Bz), dim3(128), 0, stream>>>(xcatC, keysW, encbf, xcatC + 1024);
    // new_ctx = tanh([out|ctx_att] x concat_w^T) -> bf16 straight into xcat0 ctx slot
    k_gemm<<<dim3(2, 16), blkG, 0, stream>>>(xcatC, 2048, Wcp, 64, nullptr, xcat0 + 1024, 3072, nullptr, 1);
    // actor + fused critic column
    k_gemm<<<dim3(2, 9), blkG, 0, stream>>>(xcat0 + 1024, 3072, Wap, 32, actbuf, nullptr, 576, biasA, 0);
    // softmax(pi) + value -> outputs
    k_head<<<dim3(Bz), dim3(128), 0, stream>>>(actbuf, out_logits, out_values, t);
  }
}
